// WeightedCrossEntropyLoss_54176717471798
// MI455X (gfx1250) — matrix operand order repaired and checked by reading, not compiled, not run on hardware
//
#include <hip/hip_runtime.h>
#include <hip/hip_bf16.h>

// Problem dims fixed by the reference: B=8192, V=16384, D=512.
// Workspace layout (needs ~25.4 MB):
//   Ebf     : V*D bf16 (16 MB)  -- embeddings row-normalized, bf16
//   Abf     : B*D bf16 ( 8 MB)  -- gathered target embeddings
//   topk    : B float4          -- (j1 as int, j2 as int, w1, w2)
//   rowloss : B float

typedef __attribute__((ext_vector_type(16))) __bf16 v16bf;
typedef __attribute__((ext_vector_type(8)))  float  v8f;

#define B_DIM 8192
#define V_DIM 16384
#define D_DIM 512

// ---------------------------------------------------------------------------
// K1: per-row L2 normalize embeddings and convert to bf16.
// One wave per row (512 floats -> 16 per lane), 8 waves per block.
__global__ __launch_bounds__(256) void k_normalize(const float* __restrict__ E,
                                                   __bf16* __restrict__ Ebf) {
  const int wave = threadIdx.x >> 5;
  const int lane = threadIdx.x & 31;
  const size_t row = (size_t)blockIdx.x * 8 + wave;
  const float* src = E + row * D_DIM + lane * 16;
  float v[16];
  *(float4*)&v[0]  = *(const float4*)(src + 0);
  *(float4*)&v[4]  = *(const float4*)(src + 4);
  *(float4*)&v[8]  = *(const float4*)(src + 8);
  *(float4*)&v[12] = *(const float4*)(src + 12);
  float ss = 0.f;
#pragma unroll
  for (int i = 0; i < 16; ++i) ss += v[i] * v[i];
#pragma unroll
  for (int m = 16; m >= 1; m >>= 1) ss += __shfl_xor(ss, m, 32);
  const float rn = rsqrtf(fmaxf(ss, 1e-16f));  // matches max(n_i*n_j, EPS) guard
  union { __bf16 h[16]; uint4 u[2]; } pk;
#pragma unroll
  for (int i = 0; i < 16; ++i) pk.h[i] = (__bf16)(v[i] * rn);
  uint4* dst = (uint4*)(Ebf + row * D_DIM + lane * 16);
  dst[0] = pk.u[0];
  dst[1] = pk.u[1];
}

// ---------------------------------------------------------------------------
// K2: gather target rows: Abf[b][:] = Ebf[targets[b]][:]. One wave per row.
__global__ __launch_bounds__(256) void k_gather(const __bf16* __restrict__ Ebf,
                                                const int* __restrict__ targets,
                                                __bf16* __restrict__ Abf) {
  const int wave = threadIdx.x >> 5;
  const int lane = threadIdx.x & 31;
  const size_t b = (size_t)blockIdx.x * 8 + wave;
  const size_t t = (size_t)targets[b];
  const uint4* src = (const uint4*)(Ebf + t * D_DIM);
  uint4* dst = (uint4*)(Abf + b * D_DIM);
  dst[lane] = src[lane];            // 512B
  dst[lane + 32] = src[lane + 32];  // remaining 512B
}

// ---------------------------------------------------------------------------
// K3: sims = Abf x Ebf^T via bf16 WMMA, fused relu + running top-2 per row.
// One workgroup per 16 batch rows; 8 waves split the V/16=1024 column tiles
// (2 tiles per iteration, stride 16). E (16MB bf16) is L2-resident.
__global__ __launch_bounds__(256) void k_sims_top2(const __bf16* __restrict__ Abf,
                                                   const __bf16* __restrict__ Ebf,
                                                   float4* __restrict__ topk) {
  __shared__ float sval[16][256];
  __shared__ int   sidx[16][256];
  const int tid = threadIdx.x;
  const int wave = tid >> 5;
  const int lane = tid & 31;
  const int laneLo = lane & 15;
  const int laneHi = lane >> 4;
  const int rowBase = blockIdx.x * 16;

  // per-lane running top-2 for the 8 C-matrix rows this lane holds
  float b1[8], b2[8];
  int   i1[8], i2[8];
#pragma unroll
  for (int r = 0; r < 8; ++r) { b1[r] = -1.f; b2[r] = -1.f; i1[r] = -1; i2[r] = -1; }

  // A fragment row for this lane (lane m=laneLo holds contiguous
  // K range [16*laneHi, +16) of each 32-wide K slab, matching B)
  const __bf16* Arow = Abf + (size_t)(rowBase + laneLo) * D_DIM;

  for (int tile = wave * 2; tile < V_DIM / 16; tile += 16) {
    const int col0 = tile * 16 + laneLo;
    const int col1 = col0 + 16;
    // B fragment (same contiguous k order as A: lane n=laneLo holds column n,
    // contiguous K range [16*laneHi, +16) of each 32-wide K slab)
    const __bf16* Bc0 = Ebf + (size_t)col0 * D_DIM + 16 * laneHi;
    const __bf16* Bc1 = Ebf + (size_t)col1 * D_DIM + 16 * laneHi;
    v8f c0 = {};
    v8f c1 = {};
#pragma unroll
    for (int kc = 0; kc < D_DIM / 32; ++kc) {
      const int k0 = kc * 32;
      union { v16bf v; uint4 u[2]; } a, f0, f1;
      a.u[0] = *(const uint4*)(Arow + k0 + 16 * laneHi);
      a.u[1] = *(const uint4*)(Arow + k0 + 16 * laneHi + 8);
      f0.u[0] = *(const uint4*)(Bc0 + k0);
      f0.u[1] = *(const uint4*)(Bc0 + k0 + 8);
      f1.u[0] = *(const uint4*)(Bc1 + k0);
      f1.u[1] = *(const uint4*)(Bc1 + k0 + 8);
      c0 = __builtin_amdgcn_wmma_f32_16x16x32_bf16(false, a.v, false, f0.v,
                                                   (short)0, c0, false, false);
      c1 = __builtin_amdgcn_wmma_f32_16x16x32_bf16(false, a.v, false, f1.v,
                                                   (short)0, c1, false, false);
    }
    // relu + top-2 update; columns arrive in increasing order per lane, so
    // strict '>' keeps the lower index on ties (matches jax top_k).
#pragma unroll
    for (int r = 0; r < 8; ++r) {
      float v0 = fmaxf(c0[r], 0.f);
      if (v0 > b1[r]) { b2[r] = b1[r]; i2[r] = i1[r]; b1[r] = v0; i1[r] = col0; }
      else if (v0 > b2[r]) { b2[r] = v0; i2[r] = col0; }
      float v1 = fmaxf(c1[r], 0.f);
      if (v1 > b1[r]) { b2[r] = b1[r]; i2[r] = i1[r]; b1[r] = v1; i1[r] = col1; }
      else if (v1 > b2[r]) { b2[r] = v1; i2[r] = col1; }
    }
  }

  // Stage 256 candidates per row (8 waves x 16 lanes x 2) into LDS.
#pragma unroll
  for (int r = 0; r < 8; ++r) {
    const int row = r + 8 * laneHi;
    const int slot = wave * 32 + laneLo * 2;
    sval[row][slot] = b1[r];      sidx[row][slot] = i1[r];
    sval[row][slot + 1] = b2[r];  sidx[row][slot + 1] = i2[r];
  }
  __syncthreads();

  if (tid < 16) {
    float v1 = -1.f, v2 = -1.f;
    int j1 = 0, j2 = 0;
    for (int s = 0; s < 256; ++s) {
      const float v = sval[tid][s];
      const int   j = sidx[tid][s];
      if (v > v1 || (v == v1 && (unsigned)j < (unsigned)j1)) {
        v2 = v1; j2 = j1; v1 = v; j1 = j;
      } else if (v > v2 || (v == v2 && (unsigned)j < (unsigned)j2)) {
        v2 = v; j2 = j;
      }
    }
    const float s = v1 + v2;  // >= self-sim ~ 1, never 0
    float4 o;
    o.x = __int_as_float(j1);
    o.y = __int_as_float(j2);
    o.z = v1 / s;
    o.w = v2 / s;
    topk[rowBase + tid] = o;
  }
}

// ---------------------------------------------------------------------------
// K4: single-pass streaming logsumexp per output row + per-row loss.
__global__ __launch_bounds__(256) void k_lse_loss(const float* __restrict__ out,
                                                  const float4* __restrict__ topk,
                                                  float* __restrict__ rowloss) {
  const int b = blockIdx.x;
  const float* row = out + (size_t)b * V_DIM;
  const float4* row4 = (const float4*)row;
  const int tid = threadIdx.x;
  float m = -3.402823466e38f, s = 0.f;
  for (int i = tid; i < V_DIM / 4; i += 256) {
    const float4 x = row4[i];
    const float xm = fmaxf(fmaxf(x.x, x.y), fmaxf(x.z, x.w));
    const float mn = fmaxf(m, xm);
    s = s * __expf(m - mn) + __expf(x.x - mn) + __expf(x.y - mn) +
        __expf(x.z - mn) + __expf(x.w - mn);
    m = mn;
  }
  __shared__ float sm[256];
  __shared__ float ssum[256];
  sm[tid] = m;
  ssum[tid] = s;
  __syncthreads();
  for (int st = 128; st > 0; st >>= 1) {
    if (tid < st) {
      const float m2 = sm[tid + st], s2 = ssum[tid + st];
      const float mn = fmaxf(sm[tid], m2);
      ssum[tid] = ssum[tid] * __expf(sm[tid] - mn) + s2 * __expf(m2 - mn);
      sm[tid] = mn;
    }
    __syncthreads();
  }
  if (tid == 0) {
    const float lse = sm[0] + __logf(ssum[0]);
    const float4 t = topk[b];
    const int j1 = __float_as_int(t.x);
    const int j2 = __float_as_int(t.y);
    rowloss[b] = lse - (t.z * row[j1] + t.w * row[j2]);
  }
}

// ---------------------------------------------------------------------------
// K5: deterministic fixed-order mean reduction into d_out[0].
__global__ __launch_bounds__(256) void k_reduce(const float* __restrict__ rowloss,
                                                float* __restrict__ out) {
  __shared__ float sbuf[256];
  const int tid = threadIdx.x;
  float s = 0.f;
  for (int i = tid; i < B_DIM; i += 256) s += rowloss[i];
  sbuf[tid] = s;
  __syncthreads();
  for (int st = 128; st > 0; st >>= 1) {
    if (tid < st) sbuf[tid] += sbuf[tid + st];
    __syncthreads();
  }
  if (tid == 0) out[0] = sbuf[0] * (1.f / (float)B_DIM);
}

// ---------------------------------------------------------------------------
extern "C" void kernel_launch(void* const* d_in, const int* in_sizes, int n_in,
                              void* d_out, int out_size, void* d_ws, size_t ws_size,
                              hipStream_t stream) {
  (void)in_sizes; (void)n_in; (void)out_size; (void)ws_size;
  const float* output    = (const float*)d_in[0];  // [B,V] f32
  const int*   targets   = (const int*)d_in[1];    // [B] i32
  const float* embed     = (const float*)d_in[2];  // [V,D] f32

  char* ws = (char*)d_ws;
  __bf16* Ebf = (__bf16*)ws;
  __bf16* Abf = (__bf16*)(ws + (size_t)V_DIM * D_DIM * 2);
  float4* topk = (float4*)(ws + (size_t)V_DIM * D_DIM * 2 + (size_t)B_DIM * D_DIM * 2);
  float* rowloss = (float*)((char*)topk + (size_t)B_DIM * sizeof(float4));

  k_normalize<<<V_DIM / 8, 256, 0, stream>>>(embed, Ebf);
  k_gather<<<B_DIM / 8, 256, 0, stream>>>(Ebf, targets, Abf);
  k_sims_top2<<<B_DIM / 16, 256, 0, stream>>>(Abf, Ebf, topk);
  k_lse_loss<<<B_DIM, 256, 0, stream>>>(output, topk, rowloss);
  k_reduce<<<1, 256, 0, stream>>>(rowloss, (float*)d_out);
}
